// JointNetwork_69337952026794
// MI455X (gfx1250) — compile-verified, matrix operand order checked by reading, not run on hardware
//
#include <hip/hip_runtime.h>
#include <hip/hip_bf16.h>

typedef __attribute__((ext_vector_type(16))) __bf16        v16bf;
typedef __attribute__((ext_vector_type(8)))  float         v8f;
typedef __attribute__((ext_vector_type(8)))  unsigned int  v8u;

#define JOINT 640
#define VOCAB 1024
#define NB    8
#define NT    256
#define NU    64

// gemm2 LDS layout (bytes)
#define SA_BYTES  81920            // 20 kt * 4 mt * 32 lanes * 16 bf16
#define SHE_OFF   81920            // 640 f32 = 2560 B
#define SB_OFF    84480            // 2 * 8192 B ping-pong for B tiles
#define SMEM_TOT  100864

__device__ __forceinline__ unsigned short f2bf(float f) {
  unsigned int u = __builtin_bit_cast(unsigned int, f);
  u += 0x7FFFu + ((u >> 16) & 1u);           // round-to-nearest-even
  return (unsigned short)(u >> 16);
}

// Inverse of the 16-bit WMMA K mapping (ISA 7.12.2):
// v<4 : k = half*8 + v*2 + p   (k in 0..15)
// v>=4: k = 16 + half*8 + (v-4)*2 + p
__device__ __forceinline__ void kinv(int ko, int& half, int& v, int& p) {
  if (ko < 16) { half = ko >> 3; int rem = ko & 7; v = rem >> 1;       p = rem & 1; }
  else { int q = ko - 16; half = q >> 3; int rem = q & 7; v = 4 + (rem >> 1); p = rem & 1; }
}

// fp32 row-major M x K  ->  bf16 A-fragments [mt][kt][lane][16]
__global__ void k_swizzle_A(const float* __restrict__ src,
                            unsigned short* __restrict__ dst, int M, int K) {
  int e = blockIdx.x * blockDim.x + threadIdx.x;
  if (e >= M * K) return;
  int m = e / K, k = e - m * K;
  int KT = K >> 5;
  int mt = m >> 4, r = m & 15;
  int kt = k >> 5, ko = k & 31;
  int half, v, p; kinv(ko, half, v, p);
  int lane = half * 16 + r;
  int slot = (((mt * KT + kt) * 32 + lane) << 4) + (v << 1) + p;
  dst[slot] = f2bf(src[e]);
}

// fp32 row-major K x N  ->  bf16 B-fragments [kt][nt][lane][16]
__global__ void k_swizzle_B(const float* __restrict__ src,
                            unsigned short* __restrict__ dst, int K, int N) {
  int e = blockIdx.x * blockDim.x + threadIdx.x;
  if (e >= K * N) return;
  int k = e / N, n = e - k * N;
  int NTl = N >> 4;
  int nt = n >> 4, c = n & 15;
  int kt = k >> 5, ko = k & 31;
  int half, v, p; kinv(ko, half, v, p);
  int lane = half * 16 + c;
  int slot = (((kt * NTl + nt) * 32 + lane) << 4) + (v << 1) + p;
  dst[slot] = f2bf(src[e]);
}

// GEMM1: rows 0..2047 = encoder (W1 ktiles 0..15), rows 2048..2559 = decoder (ktiles 16..31)
// HEb gets +b1 folded in.
__global__ void __launch_bounds__(256) k_gemm1(
    const unsigned short* __restrict__ fragA, const unsigned short* __restrict__ fragW1,
    const float* __restrict__ b1, float* __restrict__ HEb, float* __restrict__ HD) {
  int tid = threadIdx.x;
  int w = blockIdx.x * 8 + (tid >> 5);        // 0..6399  (160 mtiles x 40 ntiles)
  int lane = tid & 31;
  int mtile = w / 40, ntile = w - mtile * 40;
  bool isDec = (mtile >= 128);
  int ktoff = isDec ? 16 : 0;

  const v8u* A  = ((const v8u*)fragA)  + mtile * 16 * 32 + lane;
  const v8u* Bp = ((const v8u*)fragW1) + (ktoff * 40 + ntile) * 32 + lane;

  v8f c = {};
#pragma unroll
  for (int kt = 0; kt < 16; ++kt) {
    v16bf a = __builtin_bit_cast(v16bf, A[kt * 32]);
    v16bf b = __builtin_bit_cast(v16bf, Bp[kt * 40 * 32]);
    c = __builtin_amdgcn_wmma_f32_16x16x32_bf16(false, a, false, b, (short)0, c, false, false);
  }
  int n = ntile * 16 + (lane & 15);
  int rbase = (lane >> 4) * 8;                // C layout: m = r + 8*(lane/16)
  if (!isDec) {
    float bias = b1[n];
    int row0 = mtile * 16 + rbase;
#pragma unroll
    for (int r = 0; r < 8; ++r) HEb[(size_t)(row0 + r) * JOINT + n] = c[r] + bias;
  } else {
    int row0 = (mtile - 128) * 16 + rbase;
#pragma unroll
    for (int r = 0; r < 8; ++r) HD[(size_t)(row0 + r) * JOINT + n] = c[r];
  }
}

// GEMM2 (fused): block = (b, t, vt). Build h = relu(HEb[t] + HD[u]) as bf16
// A-fragments in LDS; B tiles DMA'd into LDS with async-to-LDS double buffering;
// 8 waves x (16x64 C tile) x 20 K-steps of WMMA.
__global__ void __launch_bounds__(256) k_gemm2(
    const float* __restrict__ HEb, const float* __restrict__ HD,
    const unsigned short* __restrict__ fragW2, const float* __restrict__ b2,
    float* __restrict__ out) {
  extern __shared__ char smem[];
  unsigned short* sA  = (unsigned short*)smem;            // A fragments
  float*          sHE = (float*)(smem + SHE_OFF);         // he row (biased)
  unsigned ldsBase = (unsigned)(size_t)smem;              // AS3 offset = low 32 bits of flat addr

  int tid = threadIdx.x;
  int vt = blockIdx.x & 7;
  int t  = (blockIdx.x >> 3) & 255;
  int b  = blockIdx.x >> 11;

  // ---- async-DMA helpers (ASYNCcnt-tracked, ISA 15.18.3 op 98) ----
  const char* w2base = (const char*)fragW2 + (size_t)vt * 8192;
  auto issueCopy = [&](int kt) {       // stage fragW2[kt][vt*8..vt*8+7] (8 KB contiguous)
    unsigned l = ldsBase + SB_OFF + (unsigned)((kt & 1) * 8192) + (unsigned)tid * 32u;
    const char* g = w2base + (size_t)kt * 65536 + (size_t)tid * 32;
    asm volatile("global_load_async_to_lds_b128 %0, %1, off" :: "v"(l), "v"(g) : "memory");
    asm volatile("global_load_async_to_lds_b128 %0, %1, off" :: "v"(l + 16u), "v"(g + 16) : "memory");
  };

  // ---- stage he row ----
  const float* heRow = HEb + (size_t)(b * NT + t) * JOINT;
  for (int i = tid; i < JOINT; i += 256) sHE[i] = heRow[i];
  issueCopy(0);                        // overlap B DMA with the h-tile build
  __syncthreads();

  // ---- build h tile as bf16 A-fragments in LDS ----
  const float* hdB = HD + (size_t)(b * NU) * JOINT;
  for (int e = tid; e < NU * JOINT; e += 256) {           // coalesced in j
    int u = e / JOINT, j = e - u * JOINT;
    float h = sHE[j] + hdB[(size_t)u * JOINT + j];
    h = h > 0.f ? h : 0.f;
    int kt = j >> 5, ko = j & 31;
    int half, v, p; kinv(ko, half, v, p);
    int mt = u >> 4, r = u & 15;
    int lane = half * 16 + r;
    int slot = (((kt * 4 + mt) * 32 + lane) << 4) + (v << 1) + p;
    sA[slot] = f2bf(h);
  }

  int wave = tid >> 5, lane = tid & 31;
  int mt = wave & 3, ng = wave >> 2;                      // 4 mtiles x 2 wave-groups of 4 ntiles
  const v8u* sAv = (const v8u*)sA;

  v8f a0 = {}, a1 = {}, a2 = {}, a3 = {};
  auto computeKT = [&](int kt) {
    v16bf a = __builtin_bit_cast(v16bf, sAv[(kt * 4 + mt) * 32 + lane]);
    const v8u* bp = (const v8u*)(smem + SB_OFF + (kt & 1) * 8192) + (ng * 4) * 32 + lane;
    v16bf b0 = __builtin_bit_cast(v16bf, bp[0]);
    v16bf b1f = __builtin_bit_cast(v16bf, bp[32]);
    v16bf b2f = __builtin_bit_cast(v16bf, bp[64]);
    v16bf b3f = __builtin_bit_cast(v16bf, bp[96]);
    a0 = __builtin_amdgcn_wmma_f32_16x16x32_bf16(false, a, false, b0,  (short)0, a0, false, false);
    a1 = __builtin_amdgcn_wmma_f32_16x16x32_bf16(false, a, false, b1f, (short)0, a1, false, false);
    a2 = __builtin_amdgcn_wmma_f32_16x16x32_bf16(false, a, false, b2f, (short)0, a2, false, false);
    a3 = __builtin_amdgcn_wmma_f32_16x16x32_bf16(false, a, false, b3f, (short)0, a3, false, false);
  };

  // ---- software-pipelined K loop: DMA kt+1 while computing kt ----
#pragma unroll 1
  for (int kt = 0; kt < 19; ++kt) {
    issueCopy(kt + 1);
    asm volatile("s_wait_asynccnt 1" ::: "memory");       // copy(kt) complete
    __syncthreads();                                      // buf[kt&1] + sA visible to all
    computeKT(kt);
    __syncthreads();                                      // all readers done before overwrite
  }
  asm volatile("s_wait_asynccnt 0" ::: "memory");
  __syncthreads();
  computeKT(19);

  // ---- bias + store ----
  int v0 = (vt * 8 + ng * 4) * 16 + (lane & 15);
  int rbase = (lane >> 4) * 8;
  float bias0 = b2[v0], bias1 = b2[v0 + 16], bias2 = b2[v0 + 32], bias3 = b2[v0 + 48];
  float* ob = out + ((size_t)(b * NT + t) * NU + mt * 16 + rbase) * VOCAB;
#pragma unroll
  for (int r = 0; r < 8; ++r) {
    float* orow = ob + (size_t)r * VOCAB;
    orow[v0]      = a0[r] + bias0;
    orow[v0 + 16] = a1[r] + bias1;
    orow[v0 + 32] = a2[r] + bias2;
    orow[v0 + 48] = a3[r] + bias3;
  }
}

extern "C" void kernel_launch(void* const* d_in, const int* in_sizes, int n_in,
                              void* d_out, int out_size, void* d_ws, size_t ws_size,
                              hipStream_t stream) {
  const float* enc = (const float*)d_in[0];   // 8x256x512
  const float* dec = (const float*)d_in[1];   // 8x64x512
  const float* W1  = (const float*)d_in[2];   // 1024x640
  const float* b1  = (const float*)d_in[3];   // 640
  const float* W2  = (const float*)d_in[4];   // 640x1024
  const float* b2  = (const float*)d_in[5];   // 1024
  float* out = (float*)d_out;

  char* ws = (char*)d_ws;                     // ~11.25 MB used
  unsigned short* fragA1 = (unsigned short*)(ws);             // 2,621,440 B
  unsigned short* fragW1 = (unsigned short*)(ws +  2621440);  // 1,310,720 B
  unsigned short* fragW2 = (unsigned short*)(ws +  3932160);  // 1,310,720 B
  float*          HEb    = (float*)(ws +  5242880);           // 5,242,880 B
  float*          HD     = (float*)(ws + 10485760);           // 1,310,720 B

  k_swizzle_A<<<(2048 * 512) / 256, 256, 0, stream>>>(enc, fragA1, 2048, 512);
  k_swizzle_A<<<(512 * 512) / 256, 256, 0, stream>>>(dec, fragA1 + 128 * 16 * 32 * 16, 512, 512);
  k_swizzle_B<<<(1024 * 640) / 256, 256, 0, stream>>>(W1, fragW1, 1024, 640);
  k_swizzle_B<<<(640 * 1024) / 256, 256, 0, stream>>>(W2, fragW2, 640, 1024);
  k_gemm1<<<800, 256, 0, stream>>>(fragA1, fragW1, b1, HEb, HD);
  k_gemm2<<<16384, 256, SMEM_TOT, stream>>>(HEb, HD, fragW2, b2, out);
}
